// ConditionalRoutedAttention_44590350467595
// MI455X (gfx1250) — matrix operand order repaired and checked by reading, not compiled, not run on hardware
//
#include <hip/hip_runtime.h>
#include <hip/hip_bf16.h>
#include <math.h>
#include <stdint.h>

typedef _Float16 f16;
typedef __attribute__((ext_vector_type(16))) _Float16 v16h;
typedef __attribute__((ext_vector_type(8)))  float    v8f;
typedef __attribute__((ext_vector_type(4)))  unsigned int v4u;
typedef __attribute__((ext_vector_type(4)))  int      v4i;
typedef __attribute__((ext_vector_type(8)))  int      v8i;

struct __align__(16) U16B { unsigned int x, y, z, w; };

union AFrag {
    v16h v;
    U16B u[2];
    f16  h[16];
};

#define WMMA_F16(a, b, c) \
    __builtin_amdgcn_wmma_f32_16x16x32_f16(false, (a), false, (b), (short)0, (c), false, false)

#if defined(__has_builtin)
#if __has_builtin(__builtin_amdgcn_tensor_load_to_lds) && \
    __has_builtin(__builtin_amdgcn_s_wait_tensorcnt)
#define HAVE_TDM 1
#endif
#endif

#ifdef HAVE_TDM
// ---------------------------------------------------------------------------
// Tensor Data Mover: load a 64-row x 64-f16 tile (row stride in elements)
// from global memory into LDS.  Descriptor per CDNA5 ISA:
//   group0: [1:0]=count=1 | [63:32]=lds_addr | [120:64]=global_addr | type=2
//   group1: data_size=1 (2B) | tensor_dim0=64 | tensor_dim1=8192 |
//           tile_dim0=64 | tile_dim1=64 | tensor_dim0_stride=row_stride
// Tracked on TENSORcnt; caller waits with s_wait_tensorcnt then barriers.
// ---------------------------------------------------------------------------
__device__ inline void tdm_load_tile_64x64h(const f16* g, f16* l, unsigned row_stride)
{
    const unsigned long long ga = (unsigned long long)(uintptr_t)g;
    const unsigned lds = (unsigned)(uintptr_t)l;
    v4u g0;
    g0[0] = 1u;                                                // count=1
    g0[1] = lds;                                               // lds_addr
    g0[2] = (unsigned)ga;                                      // global_addr[31:0]
    g0[3] = (unsigned)((ga >> 32) & 0x01FFFFFFull) | 0x80000000u; // ga[56:32]|type=2
    v8i g1;
    g1[0] = (int)(1u << 16);        // wg_mask=0 | data_size=1 (2 bytes)
    g1[1] = (int)(64u << 16);       // tensor_dim0 = 64 elements (bits 79:48 lo)
    g1[2] = (int)(8192u << 16);     // tensor_dim0 hi=0 | tensor_dim1 lo = 8192
    g1[3] = (int)(64u << 16);       // tensor_dim1 hi=0 | tile_dim0 = 64
    g1[4] = (int)64;                // tile_dim1 = 64 rows | tile_dim2 = 0
    g1[5] = (int)row_stride;        // tensor_dim0_stride lo (elements)
    g1[6] = 0;                      // stride hi | tensor_dim1_stride lo
    g1[7] = 0;
    const v4i gz = {0, 0, 0, 0};
#if __clang_major__ >= 23
    const v8i gz8 = {0, 0, 0, 0, 0, 0, 0, 0};
    __builtin_amdgcn_tensor_load_to_lds(g0, g1, gz, gz, gz8, 0);
#else
    __builtin_amdgcn_tensor_load_to_lds(g0, g1, gz, gz, 0);
#endif
}
#endif // HAVE_TDM

// ---------------------------------------------------------------------------
// Generic batched WMMA GEMM: C[M,N] (f32 or f16) = alpha * A[M,K] x B[K,N]
// A row-major f16.  B row-major f16 (ldb) or, if transB, B[k][n] = Bm[n*ldb+k].
// blockIdx.z decomposed as (zb, zh) with zh = z % Hz; per-axis offsets applied.
// Block = 256 threads (8 waves), 64x64 tile, K step 32.
// ---------------------------------------------------------------------------
__global__ __launch_bounds__(256) void gemm16_kernel(
    const f16* __restrict__ A, long lda,
    const f16* __restrict__ Bm, long ldb, int transB,
    float* __restrict__ Cf, f16* __restrict__ Ch, long ldc,
    int K, float alpha,
    long aob, long aoh, long bob, long boh, long cob, long coh, int Hz)
{
    const int z  = blockIdx.z;
    const int zb = z / Hz, zh = z % Hz;
    const f16* Ab = A  + (long)zb * aob + (long)zh * aoh;
    const f16* Bb = Bm + (long)zb * bob + (long)zh * boh;
    const long coff = (long)zb * cob + (long)zh * coh;

    const int m0 = blockIdx.y * 64;
    const int n0 = blockIdx.x * 64;

    __shared__ f16 As[64][32];
    __shared__ f16 Bs[32][64];

    const int tid  = threadIdx.x;
    const int lane = tid & 31;
    const int wave = tid >> 5;

    v8f acc0 = {};
    v8f acc1 = {};
    const int t0 = wave;      // tile idx in [0,16): ti = idx>>2, tj = idx&3
    const int t1 = wave + 8;

    for (int k0 = 0; k0 < K; k0 += 32) {
        // stage A tile 64x32 (one 16B chunk per thread)
        {
            const int row = tid >> 2;
            const int c8  = (tid & 3) << 3;
            *(U16B*)&As[row][c8] =
                *(const U16B*)&Ab[(long)(m0 + row) * lda + k0 + c8];
            if (k0 + 32 < K)
                __builtin_prefetch(&Ab[(long)(m0 + row) * lda + k0 + 32 + c8], 0, 3);
        }
        // stage B tile 32x64
        {
            const int row = tid >> 3;
            const int c8  = (tid & 7) << 3;
            if (!transB) {
                *(U16B*)&Bs[row][c8] =
                    *(const U16B*)&Bb[(long)(k0 + row) * ldb + n0 + c8];
                if (k0 + 32 < K)
                    __builtin_prefetch(&Bb[(long)(k0 + 32 + row) * ldb + n0 + c8], 0, 3);
            } else {
                f16 tmp[8];
                #pragma unroll
                for (int i = 0; i < 8; ++i)
                    tmp[i] = Bb[(long)(n0 + c8 + i) * ldb + k0 + row];
                #pragma unroll
                for (int i = 0; i < 8; ++i) Bs[row][c8 + i] = tmp[i];
            }
        }
        __syncthreads();

        #pragma unroll
        for (int which = 0; which < 2; ++which) {
            const int idx = which ? t1 : t0;
            const int ti = idx >> 2, tj = idx & 3;
            AFrag a, b;
            const int ar = ti * 16 + (lane & 15);
            const int kb = (lane >> 4) << 3;
            a.u[0] = *(const U16B*)&As[ar][kb];
            a.u[1] = *(const U16B*)&As[ar][kb + 16];
            #pragma unroll
            for (int t = 0; t < 16; ++t)
                b.h[t] = Bs[kb + (t & 7) + ((t >> 3) << 4)][tj * 16 + (lane & 15)];
            if (which) acc1 = WMMA_F16(a.v, b.v, acc1);
            else       acc0 = WMMA_F16(a.v, b.v, acc0);
        }
        __syncthreads();
    }

    #pragma unroll
    for (int which = 0; which < 2; ++which) {
        const int idx = which ? t1 : t0;
        const int ti = idx >> 2, tj = idx & 3;
        v8f c = which ? acc1 : acc0;
        const int col   = n0 + tj * 16 + (lane & 15);
        const int rbase = m0 + ti * 16 + ((lane >> 4) << 3);
        #pragma unroll
        for (int r = 0; r < 8; ++r) {
            const float v = c[r] * alpha;
            const long off = coff + (long)(rbase + r) * ldc + col;
            if (Ch) Ch[off] = (f16)v;
            else    Cf[off] = v;
        }
    }
}

// ---------------------------------------------------------------------------
// LayerNorm: one block per token (1024 features) -> f16
// ---------------------------------------------------------------------------
__global__ __launch_bounds__(256) void ln_kernel(
    const float* __restrict__ x, const float* __restrict__ g,
    const float* __restrict__ beta, f16* __restrict__ xl)
{
    const long t = blockIdx.x;
    const float* xr = x + t * 1024;
    __shared__ float red[256];
    const int tid = threadIdx.x;

    float s = 0.f;
    for (int i = tid; i < 1024; i += 256) s += xr[i];
    red[tid] = s; __syncthreads();
    for (int o = 128; o > 0; o >>= 1) {
        if (tid < o) red[tid] += red[tid + o];
        __syncthreads();
    }
    const float mu = red[0] * (1.f / 1024.f);
    __syncthreads();

    float s2 = 0.f;
    for (int i = tid; i < 1024; i += 256) {
        const float d = xr[i] - mu;
        s2 += d * d;
    }
    red[tid] = s2; __syncthreads();
    for (int o = 128; o > 0; o >>= 1) {
        if (tid < o) red[tid] += red[tid + o];
        __syncthreads();
    }
    const float rstd = rsqrtf(red[0] * (1.f / 1024.f) + 1e-5f);
    __syncthreads();

    for (int i = tid; i < 1024; i += 256)
        xl[t * 1024 + i] = (f16)((xr[i] - mu) * rstd * g[i] + beta[i]);
}

// ---------------------------------------------------------------------------
// Router scores: one block per token, both routers
// scores layout: [router][b*4096+n]
// ---------------------------------------------------------------------------
__global__ __launch_bounds__(256) void score_kernel(
    const float* __restrict__ x, const float* __restrict__ rtq,
    const float* __restrict__ rtkv, float* __restrict__ scores)
{
    const long t = blockIdx.x;
    const float* xr = x + t * 1024;
    __shared__ float red[256];
    const int tid = threadIdx.x;

    float s = 0.f;
    for (int i = tid; i < 1024; i += 256) s += xr[i] * rtq[i];
    red[tid] = s; __syncthreads();
    for (int o = 128; o > 0; o >>= 1) {
        if (tid < o) red[tid] += red[tid + o];
        __syncthreads();
    }
    if (tid == 0) scores[t] = red[0];
    __syncthreads();

    s = 0.f;
    for (int i = tid; i < 1024; i += 256) s += xr[i] * rtkv[i];
    red[tid] = s; __syncthreads();
    for (int o = 128; o > 0; o >>= 1) {
        if (tid < o) red[tid] += red[tid + o];
        __syncthreads();
    }
    if (tid == 0) scores[8192 + t] = red[0];
}

// ---------------------------------------------------------------------------
// Per-(router,batch) bitonic full sort of 4096 scores; keep top-1024 indices.
// blockIdx.x = router*2 + batch. 1024 threads. 32KB LDS.
// ---------------------------------------------------------------------------
__global__ __launch_bounds__(1024) void route_sort_kernel(
    const float* __restrict__ scores, int* __restrict__ sel)
{
    const int b = blockIdx.x & 1;
    const int r = blockIdx.x >> 1;
    const float* sc = scores + (long)r * 8192 + (long)b * 4096;

    __shared__ float key[4096];
    __shared__ int   vid[4096];
    const int tid = threadIdx.x;

    for (int i = tid; i < 4096; i += 1024) { key[i] = sc[i]; vid[i] = i; }
    __syncthreads();

    for (int k = 2; k <= 4096; k <<= 1) {
        for (int j = k >> 1; j > 0; j >>= 1) {
            for (int i = tid; i < 4096; i += 1024) {
                const int ixj = i ^ j;
                if (ixj > i) {
                    const bool up = ((i & k) == 0);
                    const bool doswap = up ? (key[i] > key[ixj])
                                           : (key[i] < key[ixj]);
                    if (doswap) {
                        float tk = key[i]; key[i] = key[ixj]; key[ixj] = tk;
                        int   tv = vid[i]; vid[i] = vid[ixj]; vid[ixj] = tv;
                    }
                }
            }
            __syncthreads();
        }
    }
    for (int rr = tid; rr < 1024; rr += 1024)
        sel[((long)r * 2 + b) * 1024 + rr] = vid[3072 + rr];
}

// ---------------------------------------------------------------------------
// Gather routed tokens + RMS norm * sqrt(D) * gamma -> f16
// blockIdx: (row 0..1023, batch, router)
// ---------------------------------------------------------------------------
__global__ __launch_bounds__(256) void gather_rms_kernel(
    const float* __restrict__ x, const float* __restrict__ gamma,
    const int* __restrict__ sel, f16* __restrict__ rq, f16* __restrict__ rkv)
{
    const int row = blockIdx.x, b = blockIdx.y, r = blockIdx.z;
    const int tok = sel[((long)r * 2 + b) * 1024 + row];
    const float* xr = x + ((long)b * 4096 + tok) * 1024;

    __shared__ float red[256];
    const int tid = threadIdx.x;
    float s2 = 0.f;
    for (int i = tid; i < 1024; i += 256) { const float v = xr[i]; s2 += v * v; }
    red[tid] = s2; __syncthreads();
    for (int o = 128; o > 0; o >>= 1) {
        if (tid < o) red[tid] += red[tid + o];
        __syncthreads();
    }
    float nrm = sqrtf(red[0]);
    nrm = fmaxf(nrm, 1e-12f);
    const float scl = 32.0f / nrm;   // sqrt(1024) / ||x||

    f16* out = (r == 0 ? rq : rkv) + ((long)b * 1024 + row) * 1024;
    for (int i = tid; i < 1024; i += 256)
        out[i] = (f16)(xr[i] * scl * gamma[i]);
}

// ---------------------------------------------------------------------------
// Fused local-window attention. grid = (64 windows, 8 heads, 2 batches).
// qkv layout: [token][1536] with q at col h*64, k at 512+h*64, v at 1024+h*64.
// LDS arena 56KB: Q[64][64] | K[192][64] | S/P f16 [64][192]; V aliases Q+K.
// Q/K/V tiles staged by the Tensor Data Mover (wave 0 issues descriptors,
// waits TENSORcnt); invalid look-around windows zero-filled by threads.
// ---------------------------------------------------------------------------
__global__ __launch_bounds__(256) void local_attn_kernel(
    const f16* __restrict__ qkv, f16* __restrict__ lo)
{
    const int wi = blockIdx.x, h = blockIdx.y, b = blockIdx.z;
    __shared__ __align__(16) char smem[57344];
    f16 (*Qs)[64]  = (f16(*)[64]) smem;             // 0   .. 8K
    f16 (*Ks)[64]  = (f16(*)[64])(smem + 8192);     // 8K  .. 32K (192 rows)
    f16 (*Ss)[192] = (f16(*)[192])(smem + 32768);   // 32K .. 56K
    f16 (*Vs)[64]  = (f16(*)[64]) smem;             // aliases Q+K after S

    const int tid = threadIdx.x, lane = tid & 31, wave = tid >> 5;
    const long tb = (long)b * 4096 + (long)wi * 64;

#ifdef HAVE_TDM
    // --- TDM staging of Q and the three K windows ---
    if (wave == 0) {
        tdm_load_tile_64x64h(&qkv[tb * 1536 + h * 64], &Qs[0][0], 1536u);
        #pragma unroll
        for (int s = 0; s < 3; ++s) {
            const int sw = wi + s - 1;
            if (sw >= 0 && sw < 64)
                tdm_load_tile_64x64h(
                    &qkv[((long)b * 4096 + (long)sw * 64) * 1536 + 512 + h * 64],
                    &Ks[s * 64][0], 1536u);
        }
    }
    // zero-fill rows of invalid windows (disjoint from TDM-written rows)
    for (int u = tid; u < 1536; u += 256) {
        const int row = u >> 3, c8 = (u & 7) << 3;
        const int sw = wi + (row >> 6) - 1;
        if (sw < 0 || sw >= 64) {
            const U16B zz = {0u, 0u, 0u, 0u};
            *(U16B*)&Ks[row][c8] = zz;
        }
    }
    if (wave == 0) __builtin_amdgcn_s_wait_tensorcnt(0);
#else
    for (int u = tid; u < 512; u += 256) {
        const int row = u >> 3, c8 = (u & 7) << 3;
        *(U16B*)&Qs[row][c8] =
            *(const U16B*)&qkv[(tb + row) * 1536 + h * 64 + c8];
    }
    for (int u = tid; u < 1536; u += 256) {
        const int row = u >> 3, c8 = (u & 7) << 3;
        const int sw = wi + (row >> 6) - 1, jj = row & 63;
        U16B val = {0u, 0u, 0u, 0u};
        if (sw >= 0 && sw < 64)
            val = *(const U16B*)
                &qkv[((long)b * 4096 + (long)sw * 64 + jj) * 1536 + 512 + h * 64 + c8];
        *(U16B*)&Ks[row][c8] = val;
    }
#endif
    __syncthreads();

    // S = scale * Q K^T with look-around mask; 48 tiles over 8 waves
    for (int idx = wave; idx < 48; idx += 8) {
        const int ti = idx / 12, tj = idx % 12;
        v8f c = {};
        #pragma unroll
        for (int kk = 0; kk < 64; kk += 32) {
            AFrag a, bfr;
            const int ar = ti * 16 + (lane & 15);
            const int kb = kk + ((lane >> 4) << 3);
            a.u[0] = *(const U16B*)&Qs[ar][kb];
            a.u[1] = *(const U16B*)&Qs[ar][kb + 16];
            const int krow = tj * 16 + (lane & 15);
            bfr.u[0] = *(const U16B*)&Ks[krow][kb];
            bfr.u[1] = *(const U16B*)&Ks[krow][kb + 16];
            c = WMMA_F16(a.v, bfr.v, c);
        }
        const int col = tj * 16 + (lane & 15);
        const int sw  = wi + (col >> 6) - 1;
        const bool valid = (sw >= 0) && (sw < 64);
        const int rbase = ti * 16 + ((lane >> 4) << 3);
        #pragma unroll
        for (int r = 0; r < 8; ++r)
            Ss[rbase + r][col] = valid ? (f16)(c[r] * 0.125f) : (f16)(-60000.0f);
    }
    __syncthreads();

    // stage V (into dead Q+K region) while threads 0..63 run row softmax
#ifdef HAVE_TDM
    if (wave == 0) {
        #pragma unroll
        for (int s = 0; s < 3; ++s) {
            const int sw = wi + s - 1;
            if (sw >= 0 && sw < 64)
                tdm_load_tile_64x64h(
                    &qkv[((long)b * 4096 + (long)sw * 64) * 1536 + 1024 + h * 64],
                    &Vs[s * 64][0], 1536u);
        }
    }
    for (int u = tid; u < 1536; u += 256) {
        const int row = u >> 3, c8 = (u & 7) << 3;
        const int sw = wi + (row >> 6) - 1;
        if (sw < 0 || sw >= 64) {
            const U16B zz = {0u, 0u, 0u, 0u};
            *(U16B*)&Vs[row][c8] = zz;
        }
    }
#else
    for (int u = tid; u < 1536; u += 256) {
        const int row = u >> 3, c8 = (u & 7) << 3;
        const int sw = wi + (row >> 6) - 1, jj = row & 63;
        U16B val = {0u, 0u, 0u, 0u};
        if (sw >= 0 && sw < 64)
            val = *(const U16B*)
                &qkv[((long)b * 4096 + (long)sw * 64 + jj) * 1536 + 1024 + h * 64 + c8];
        *(U16B*)&Vs[row][c8] = val;
    }
#endif
    if (tid < 64) {
        const int row = tid;
        float mx = -1e30f;
        for (int j = 0; j < 192; ++j) mx = fmaxf(mx, (float)Ss[row][j]);
        float sum = 0.f;
        for (int j = 0; j < 192; ++j) sum += __expf((float)Ss[row][j] - mx);
        const float inv = 1.f / sum;
        for (int j = 0; j < 192; ++j) {
            const float p = __expf((float)Ss[row][j] - mx) * inv;
            Ss[row][j] = (f16)p;     // overwrite S with P in place
        }
    }
#ifdef HAVE_TDM
    if (wave == 0) __builtin_amdgcn_s_wait_tensorcnt(0);
#endif
    __syncthreads();

    // O = P V : 16 tiles over 8 waves, K=192
    for (int idx = wave; idx < 16; idx += 8) {
        const int ti = idx >> 2, td = idx & 3;
        v8f c = {};
        #pragma unroll
        for (int kk = 0; kk < 192; kk += 32) {
            AFrag a, bfr;
            const int ar = ti * 16 + (lane & 15);
            const int kb = kk + ((lane >> 4) << 3);
            a.u[0] = *(const U16B*)&Ss[ar][kb];
            a.u[1] = *(const U16B*)&Ss[ar][kb + 16];
            #pragma unroll
            for (int t = 0; t < 16; ++t)
                bfr.h[t] = Vs[kb + (t & 7) + ((t >> 3) << 4)][td * 16 + (lane & 15)];
            c = WMMA_F16(a.v, bfr.v, c);
        }
        const int col   = td * 16 + (lane & 15);
        const int rbase = ti * 16 + ((lane >> 4) << 3);
        #pragma unroll
        for (int r = 0; r < 8; ++r)
            lo[(tb + rbase + r) * 512 + h * 64 + col] = (f16)c[r];
    }
}

// ---------------------------------------------------------------------------
// Heavy attention row softmax: one block per row (1024 cols), f32 S -> f16 P
// ---------------------------------------------------------------------------
__global__ __launch_bounds__(256) void softmax_heavy_kernel(
    const float* __restrict__ S, f16* __restrict__ P)
{
    const long row = blockIdx.x;
    const float* sr = S + row * 1024;
    __shared__ float red[256];
    const int tid = threadIdx.x;

    float mx = -1e30f;
    for (int i = tid; i < 1024; i += 256) mx = fmaxf(mx, sr[i]);
    red[tid] = mx; __syncthreads();
    for (int o = 128; o > 0; o >>= 1) {
        if (tid < o) red[tid] = fmaxf(red[tid], red[tid + o]);
        __syncthreads();
    }
    const float m = red[0];
    __syncthreads();

    float sum = 0.f;
    for (int i = tid; i < 1024; i += 256) sum += __expf(sr[i] - m);
    red[tid] = sum; __syncthreads();
    for (int o = 128; o > 0; o >>= 1) {
        if (tid < o) red[tid] += red[tid + o];
        __syncthreads();
    }
    const float inv = 1.f / red[0];

    for (int i = tid; i < 1024; i += 256)
        P[row * 1024 + i] = (f16)(__expf(sr[i] - m) * inv);
}

// ---------------------------------------------------------------------------
// Scatter-add heavy output rows onto d_out at routed token positions
// ---------------------------------------------------------------------------
__global__ __launch_bounds__(256) void scatter_add_kernel(
    const float* __restrict__ hout, const int* __restrict__ selq,
    float* __restrict__ out)
{
    const int rowg = blockIdx.x;
    const int b = rowg >> 10, rr = rowg & 1023;
    const int tok = selq[(long)b * 1024 + rr];
    float* o = out + ((long)b * 4096 + tok) * 1024;
    const float* hi = hout + (long)rowg * 1024;
    for (int i = threadIdx.x; i < 1024; i += 256) o[i] += hi[i];
}

// ---------------------------------------------------------------------------
// f32 -> f16 elementwise (weight conversion)
// ---------------------------------------------------------------------------
__global__ __launch_bounds__(256) void f32_to_f16_kernel(
    const float* __restrict__ in, f16* __restrict__ out, int n)
{
    const int i = blockIdx.x * 256 + threadIdx.x;
    if (i < n) out[i] = (f16)in[i];
}

// ---------------------------------------------------------------------------
extern "C" void kernel_launch(void* const* d_in, const int* in_sizes, int n_in,
                              void* d_out, int out_size, void* d_ws, size_t ws_size,
                              hipStream_t stream)
{
    const float* x       = (const float*)d_in[0];
    const float* gamma_h = (const float*)d_in[1];
    const float* wq_h    = (const float*)d_in[2];
    const float* wkv_h   = (const float*)d_in[3];
    const float* wo_h    = (const float*)d_in[4];
    const float* ln_g    = (const float*)d_in[5];
    const float* ln_b    = (const float*)d_in[6];
    const float* wqkv_l  = (const float*)d_in[7];
    const float* wo_l    = (const float*)d_in[8];
    const float* rt_q    = (const float*)d_in[9];
    const float* rt_kv   = (const float*)d_in[10];
    float* out = (float*)d_out;

    char* base = (char*)d_ws;
    size_t off = 0;
    auto alloc = [&](size_t bytes) -> void* {
        void* p = base + off;
        off = (off + bytes + 255) & ~(size_t)255;
        return p;
    };

    f16*   xl16    = (f16*)  alloc((size_t)8192 * 1024 * 2);
    f16*   qkv16   = (f16*)  alloc((size_t)8192 * 1536 * 2);
    f16*   lo16    = (f16*)  alloc((size_t)8192 * 512  * 2);
    f16*   wqkv16  = (f16*)  alloc((size_t)1024 * 1536 * 2);
    f16*   wol16   = (f16*)  alloc((size_t)512  * 1024 * 2);
    f16*   wqh16   = (f16*)  alloc((size_t)1024 * 512  * 2);
    f16*   wkvh16  = (f16*)  alloc((size_t)1024 * 1024 * 2);
    f16*   woh16   = (f16*)  alloc((size_t)512  * 1024 * 2);
    float* scores  = (float*)alloc((size_t)2 * 8192 * 4);
    int*   selidx  = (int*)  alloc((size_t)2 * 2 * 1024 * 4);
    f16*   rq16    = (f16*)  alloc((size_t)2048 * 1024 * 2);
    f16*   rkv16   = (f16*)  alloc((size_t)2048 * 1024 * 2);
    f16*   hq16    = (f16*)  alloc((size_t)2048 * 512  * 2);
    f16*   hkv16   = (f16*)  alloc((size_t)2048 * 1024 * 2);
    float* Sbuf    = (float*)alloc((size_t)16 * 1024 * 1024 * 4);
    f16*   P16     = (f16*)  alloc((size_t)16 * 1024 * 1024 * 2);
    f16*   ho16    = (f16*)  alloc((size_t)2048 * 512  * 2);
    float* hout    = (float*)alloc((size_t)2048 * 1024 * 4);
    (void)ws_size; (void)in_sizes; (void)n_in; (void)out_size;

    auto gemm = [&](const f16* A, long lda, const f16* Bm, long ldb, int tB,
                    float* Cf, f16* Ch, long ldc, int M, int N, int K,
                    float alpha, long aob, long aoh, long bob, long boh,
                    long cob, long coh, int Hz, int Z) {
        dim3 g((unsigned)(N / 64), (unsigned)(M / 64), (unsigned)Z);
        gemm16_kernel<<<g, dim3(256), 0, stream>>>(
            A, lda, Bm, ldb, tB, Cf, Ch, ldc, K, alpha,
            aob, aoh, bob, boh, cob, coh, Hz);
    };

    // --- weight conversions to f16 ---
    f32_to_f16_kernel<<<(1024 * 1536 + 255) / 256, 256, 0, stream>>>(wqkv_l, wqkv16, 1024 * 1536);
    f32_to_f16_kernel<<<(512  * 1024 + 255) / 256, 256, 0, stream>>>(wo_l,   wol16,  512 * 1024);
    f32_to_f16_kernel<<<(1024 * 512  + 255) / 256, 256, 0, stream>>>(wq_h,   wqh16,  1024 * 512);
    f32_to_f16_kernel<<<(1024 * 1024 + 255) / 256, 256, 0, stream>>>(wkv_h,  wkvh16, 1024 * 1024);
    f32_to_f16_kernel<<<(512  * 1024 + 255) / 256, 256, 0, stream>>>(wo_h,   woh16,  512 * 1024);

    // --- light branch ---
    ln_kernel<<<8192, 256, 0, stream>>>(x, ln_g, ln_b, xl16);
    // qkv = xl @ wqkv_l  (8192 x 1536 x 1024)
    gemm(xl16, 1024, wqkv16, 1536, 0, nullptr, qkv16, 1536,
         8192, 1536, 1024, 1.0f, 0, 0, 0, 0, 0, 0, 1, 1);
    // fused windowed attention (TDM-staged tiles)
    local_attn_kernel<<<dim3(64, 8, 2), 256, 0, stream>>>(qkv16, lo16);
    // light_out = lo @ wo_l -> d_out (f32)
    gemm(lo16, 512, wol16, 1024, 0, out, nullptr, 1024,
         8192, 1024, 512, 1.0f, 0, 0, 0, 0, 0, 0, 1, 1);

    // --- routing (forward gates == 1.0; only indices matter) ---
    score_kernel<<<8192, 256, 0, stream>>>(x, rt_q, rt_kv, scores);
    route_sort_kernel<<<4, 1024, 0, stream>>>(scores, selidx);
    gather_rms_kernel<<<dim3(1024, 2, 2), 256, 0, stream>>>(x, gamma_h, selidx, rq16, rkv16);

    // --- heavy branch ---
    // hq = rms(rq) @ wq_h  (2048 x 512 x 1024)
    gemm(rq16, 1024, wqh16, 512, 0, nullptr, hq16, 512,
         2048, 512, 1024, 1.0f, 0, 0, 0, 0, 0, 0, 1, 1);
    // hk|hv = rms(rkv) @ wkv_h  (2048 x 1024 x 1024)
    gemm(rkv16, 1024, wkvh16, 1024, 0, nullptr, hkv16, 1024,
         2048, 1024, 1024, 1.0f, 0, 0, 0, 0, 0, 0, 1, 1);
    // S[b,h] = scale * hq hk^T  (batched z = b*8+h; 1024 x 1024 x 64)
    gemm(hq16, 512, hkv16, 1024, 1, Sbuf, nullptr, 1024,
         1024, 1024, 64, 0.125f,
         /*aob*/ (long)1024 * 512, /*aoh*/ 64,
         /*bob*/ (long)1024 * 1024, /*boh*/ 64,
         /*cob*/ (long)8 * 1024 * 1024, /*coh*/ (long)1024 * 1024,
         /*Hz*/ 8, /*Z*/ 16);
    softmax_heavy_kernel<<<16384, 256, 0, stream>>>(Sbuf, P16);
    // ho[b,h] = P @ hv  (1024 x 64 x 1024)
    gemm(P16, 1024, hkv16 + 512, 1024, 0, nullptr, ho16, 512,
         1024, 64, 1024, 1.0f,
         /*aob*/ (long)8 * 1024 * 1024, /*aoh*/ (long)1024 * 1024,
         /*bob*/ (long)1024 * 1024, /*boh*/ 64,
         /*cob*/ (long)1024 * 512, /*coh*/ 64,
         /*Hz*/ 8, /*Z*/ 16);
    // hout = ho @ wo_h  (2048 x 1024 x 512)
    gemm(ho16, 512, woh16, 1024, 0, hout, nullptr, 1024,
         2048, 1024, 512, 1.0f, 0, 0, 0, 0, 0, 0, 1, 1);
    // scatter heavy rows onto light output (selected rows are unique)
    scatter_add_kernel<<<2048, 256, 0, stream>>>(hout, selidx, out);
}
